// EmbodiedCTRNN_50723563766279
// MI455X (gfx1250) — matrix operand order repaired and checked by reading, not compiled, not run on hardware
//
#include <hip/hip_runtime.h>
#include <hip/hip_bf16.h>

typedef __attribute__((ext_vector_type(16))) __bf16 v16bf;
typedef __attribute__((ext_vector_type(8)))  __bf16 v8bf;
typedef __attribute__((ext_vector_type(8)))  float  v8f;

#define T_STEPS  512
#define BATCH    256
#define IN_DIM   128
#define H_DIM    512
#define BODY_DIM 64
#define KCAT     (IN_DIM + H_DIM + BODY_DIM)   // 704
#define ALPHA_V  0.1f

#define ROWS_PER_WG 16
#define NUM_WG      (BATCH / ROWS_PER_WG)      // 16
#define THREADS     256                        // 8 wave32 waves

// Workspace layout (needs 788,480 bytes):
//   [0)           Wcat bf16 [H_DIM][KCAT]      = 720,896 B  ( W_i2h | W_h2h | W_b2h )
//   [720896)      Wb   bf16 [BODY_DIM][H_DIM]  =  65,536 B
//   [786432)      bias_h f32 [H_DIM]           =   2,048 B  ( b_i2h + b_h2h + b_b2h )
#define WS_WCAT_OFF 0
#define WS_WB_OFF   720896
#define WS_BIAS_OFF 786432

// ---------------------------------------------------------------------------
// Prep: pack weights to bf16 (concatenated K layout) + fuse hidden biases.
// ---------------------------------------------------------------------------
__global__ void ctrnn_prep_kernel(const float* __restrict__ W_i2h,
                                  const float* __restrict__ b_i2h,
                                  const float* __restrict__ W_h2h,
                                  const float* __restrict__ b_h2h,
                                  const float* __restrict__ W_b2h,
                                  const float* __restrict__ b_b2h,
                                  const float* __restrict__ W_h2b,
                                  __hip_bfloat16* __restrict__ Wcat,
                                  __hip_bfloat16* __restrict__ Wb,
                                  float* __restrict__ bias_h) {
  const int stride = gridDim.x * blockDim.x;
  const int tid    = blockIdx.x * blockDim.x + threadIdx.x;

  for (int i = tid; i < H_DIM * KCAT; i += stride) {
    const int n = i / KCAT;
    const int k = i % KCAT;
    float v;
    if (k < IN_DIM)                v = W_i2h[n * IN_DIM + k];
    else if (k < IN_DIM + H_DIM)   v = W_h2h[n * H_DIM + (k - IN_DIM)];
    else                           v = W_b2h[n * BODY_DIM + (k - IN_DIM - H_DIM)];
    Wcat[i] = __float2bfloat16(v);
  }
  for (int i = tid; i < BODY_DIM * H_DIM; i += stride)
    Wb[i] = __float2bfloat16(W_h2b[i]);
  for (int i = tid; i < H_DIM; i += stride)
    bias_h[i] = b_i2h[i] + b_h2h[i] + b_b2h[i];
}

// ---------------------------------------------------------------------------
// A-fragment (bf16 16x32): lanes 0-15 hold M=lane, K chunks {0..7, 16..23};
// lanes 16-31 hold M=lane-16, K chunks {8..15, 24..31}.
// ---------------------------------------------------------------------------
__device__ __forceinline__ v16bf load_a_frag(const __hip_bfloat16* __restrict__ row,
                                             int kk, int hlf) {
  const v8bf c1 = *reinterpret_cast<const v8bf*>(row + kk + hlf * 8);
  const v8bf c2 = *reinterpret_cast<const v8bf*>(row + kk + 16 + hlf * 8);
  v16bf a;
#pragma unroll
  for (int i = 0; i < 8; ++i) { a[i] = c1[i]; a[8 + i] = c2[i]; }
  return a;
}

// B-fragment (bf16 32x16): lanes 0-15 hold N=lane, K chunks {0..7, 16..23};
// lanes 16-31 hold N=lane-16, K chunks {8..15, 24..31}. Two 16-byte loads.
__device__ __forceinline__ v16bf load_b_frag(const __hip_bfloat16* __restrict__ wrow,
                                             int kk, int hlf) {
  const v8bf c1 = *reinterpret_cast<const v8bf*>(wrow + kk + hlf * 8);
  const v8bf c2 = *reinterpret_cast<const v8bf*>(wrow + kk + 16 + hlf * 8);
  v16bf b;
#pragma unroll
  for (int i = 0; i < 8; ++i) { b[i] = c1[i]; b[8 + i] = c2[i]; }
  return b;
}

__device__ __forceinline__ v8f wmma_bf16(v16bf a, v16bf b, v8f c) {
  return __builtin_amdgcn_wmma_f32_16x16x32_bf16(false, a, false, b,
                                                 (short)0, c, false, false);
}

// ---------------------------------------------------------------------------
// Persistent scan kernel: 16 WGs x 16 batch rows, 512 sequential steps.
// ---------------------------------------------------------------------------
__global__ __launch_bounds__(THREADS)
void ctrnn_scan_kernel(const float* __restrict__ x,
                       const __hip_bfloat16* __restrict__ Wcat,
                       const __hip_bfloat16* __restrict__ Wb,
                       const float* __restrict__ bias_h,
                       const float* __restrict__ b_h2b,
                       const float* __restrict__ body_mask,
                       float* __restrict__ hidden_out,
                       float* __restrict__ body_out,
                       float* __restrict__ h_fin) {
  __shared__ __align__(32) __hip_bfloat16 act[ROWS_PER_WG][KCAT]; // [x|h|b] bf16
  __shared__ __align__(32) float h32[ROWS_PER_WG][H_DIM];         // fp32 hidden state
  __shared__ __align__(32) float b32[ROWS_PER_WG][BODY_DIM];      // fp32 body state

  const int batch0 = blockIdx.x * ROWS_PER_WG;
  const int lane   = threadIdx.x & 31;
  // Scalarized wave index: keeps all wave-level control flow on the SALU so
  // no EXEC-mask save/restore ever wraps a WMMA.
  const int wv     = __builtin_amdgcn_readfirstlane(threadIdx.x >> 5); // 0..7
  const int hlf    = lane >> 4;          // 0 or 1
  const int lcol   = lane & 15;          // column within tile

  // ---- init state: h = 0, b = 0 (both fp32 copies and bf16 act slots) ----
  for (int i = threadIdx.x; i < ROWS_PER_WG * H_DIM; i += THREADS) {
    h32[i >> 9][i & (H_DIM - 1)] = 0.0f;
    act[i >> 9][IN_DIM + (i & (H_DIM - 1))] = __float2bfloat16(0.0f);
  }
  for (int i = threadIdx.x; i < ROWS_PER_WG * BODY_DIM; i += THREADS) {
    b32[i >> 6][i & (BODY_DIM - 1)] = 0.0f;
    act[i >> 6][IN_DIM + H_DIM + (i & (BODY_DIM - 1))] = __float2bfloat16(0.0f);
  }

  // ---- per-lane constants: bias for the 4 hidden tiles, body bias/mask ----
  float biasv[4];
#pragma unroll
  for (int j = 0; j < 4; ++j) biasv[j] = bias_h[wv * 64 + j * 16 + lcol];
  float bbias = 0.0f, bmask = 0.0f;
  if (wv < 4) {
    bbias = b_h2b[wv * 16 + lcol];
    bmask = body_mask[wv * 16 + lcol];
  }

  // Per-lane weight row pointers (loop-invariant).
  const __hip_bfloat16* __restrict__ wrow0 = Wcat + (size_t)(wv * 64 +  0 + lcol) * KCAT;
  const __hip_bfloat16* __restrict__ wrow1 = Wcat + (size_t)(wv * 64 + 16 + lcol) * KCAT;
  const __hip_bfloat16* __restrict__ wrow2 = Wcat + (size_t)(wv * 64 + 32 + lcol) * KCAT;
  const __hip_bfloat16* __restrict__ wrow3 = Wcat + (size_t)(wv * 64 + 48 + lcol) * KCAT;
  const __hip_bfloat16* __restrict__ wbrow = Wb + (size_t)(wv * 16 + lcol) * H_DIM;

  __syncthreads();

  for (int t = 0; t < T_STEPS; ++t) {
    // ---- stage x[t] tile into act[:, 0:128] as bf16 (x is read-once: NT) ----
    for (int i = threadIdx.x; i < ROWS_PER_WG * IN_DIM; i += THREADS) {
      const int r = i >> 7, c = i & (IN_DIM - 1);
      const float xv = __builtin_nontemporal_load(
          &x[((size_t)t * BATCH + batch0 + r) * IN_DIM + c]);
      act[r][c] = __float2bfloat16(xv);
    }
    __syncthreads();  // act fully written (x this iter, h/b last iter)

    // ---- GEMMs on OLD state: pre = act @ Wcat^T ; accB = h_old @ Wb^T ----
    v8f a0 = {}, a1 = {}, a2 = {}, a3 = {};
    v8f accB = {};
    const __hip_bfloat16* __restrict__ arow = &act[lcol][0];

    // Segment 1: K = [0, 128)  (input projection part, hidden only)
#pragma unroll
    for (int kk = 0; kk < IN_DIM; kk += 32) {
      const v16bf a = load_a_frag(arow, kk, hlf);
      a0 = wmma_bf16(a, load_b_frag(wrow0, kk, hlf), a0);
      a1 = wmma_bf16(a, load_b_frag(wrow1, kk, hlf), a1);
      a2 = wmma_bf16(a, load_b_frag(wrow2, kk, hlf), a2);
      a3 = wmma_bf16(a, load_b_frag(wrow3, kk, hlf), a3);
    }
    // Segment 2: K = [128, 640)  (h part; waves 0-3 also run the body GEMM).
    // Branch chosen once on an SGPR -> two branch-free inner loops.
    if (wv < 4) {
      for (int kk = IN_DIM; kk < IN_DIM + H_DIM; kk += 32) {
        const v16bf a = load_a_frag(arow, kk, hlf);
        a0 = wmma_bf16(a, load_b_frag(wrow0, kk, hlf), a0);
        a1 = wmma_bf16(a, load_b_frag(wrow1, kk, hlf), a1);
        a2 = wmma_bf16(a, load_b_frag(wrow2, kk, hlf), a2);
        a3 = wmma_bf16(a, load_b_frag(wrow3, kk, hlf), a3);
        accB = wmma_bf16(a, load_b_frag(wbrow, kk - IN_DIM, hlf), accB);
      }
    } else {
      for (int kk = IN_DIM; kk < IN_DIM + H_DIM; kk += 32) {
        const v16bf a = load_a_frag(arow, kk, hlf);
        a0 = wmma_bf16(a, load_b_frag(wrow0, kk, hlf), a0);
        a1 = wmma_bf16(a, load_b_frag(wrow1, kk, hlf), a1);
        a2 = wmma_bf16(a, load_b_frag(wrow2, kk, hlf), a2);
        a3 = wmma_bf16(a, load_b_frag(wrow3, kk, hlf), a3);
      }
    }
    // Segment 3: K = [640, 704)  (body-state part, hidden only)
#pragma unroll
    for (int kk = IN_DIM + H_DIM; kk < KCAT; kk += 32) {
      const v16bf a = load_a_frag(arow, kk, hlf);
      a0 = wmma_bf16(a, load_b_frag(wrow0, kk, hlf), a0);
      a1 = wmma_bf16(a, load_b_frag(wrow1, kk, hlf), a1);
      a2 = wmma_bf16(a, load_b_frag(wrow2, kk, hlf), a2);
      a3 = wmma_bf16(a, load_b_frag(wrow3, kk, hlf), a3);
    }
    __syncthreads();  // all reads of act / old state complete

    // ---- hidden update: h' = (1-a)h + a*relu(pre) ----
    v8f accH[4] = {a0, a1, a2, a3};
#pragma unroll
    for (int j = 0; j < 4; ++j) {
      const int n = wv * 64 + j * 16 + lcol;
#pragma unroll
      for (int r = 0; r < 8; ++r) {
        const int m   = r + hlf * 8;
        const float p = accH[j][r] + biasv[j];
        const float hnew =
            h32[m][n] * (1.0f - ALPHA_V) + ALPHA_V * fmaxf(p, 0.0f);
        h32[m][n] = hnew;
        act[m][IN_DIM + n] = __float2bfloat16(hnew);
        __builtin_nontemporal_store(
            hnew, &hidden_out[(size_t)t * BATCH * H_DIM +
                              (size_t)(batch0 + m) * H_DIM + n]);
      }
    }

    // ---- body update: b' = b*mask + a*(h_old @ Wb^T + b_h2b) ----
    if (wv < 4) {
      const int n = wv * 16 + lcol;
#pragma unroll
      for (int r = 0; r < 8; ++r) {
        const int m = r + hlf * 8;
        const float bnew = b32[m][n] * bmask + ALPHA_V * (accB[r] + bbias);
        b32[m][n] = bnew;
        act[m][IN_DIM + H_DIM + n] = __float2bfloat16(bnew);
        __builtin_nontemporal_store(
            bnew, &body_out[(size_t)t * BATCH * BODY_DIM +
                            (size_t)(batch0 + m) * BODY_DIM + n]);
      }
    }
    // no barrier needed here: next x-stage writes a disjoint act region and
    // the barrier after it orders these writes before the next GEMM reads.
  }

  __syncthreads();
  // ---- final hidden state ----
  for (int i = threadIdx.x; i < ROWS_PER_WG * H_DIM; i += THREADS) {
    const int r = i >> 9, n = i & (H_DIM - 1);
    h_fin[(size_t)(batch0 + r) * H_DIM + n] = h32[r][n];
  }
}

// ---------------------------------------------------------------------------
extern "C" void kernel_launch(void* const* d_in, const int* in_sizes, int n_in,
                              void* d_out, int out_size, void* d_ws, size_t ws_size,
                              hipStream_t stream) {
  const float* x         = (const float*)d_in[0];
  const float* W_i2h     = (const float*)d_in[1];
  const float* b_i2h     = (const float*)d_in[2];
  const float* W_h2h     = (const float*)d_in[3];
  const float* b_h2h     = (const float*)d_in[4];
  const float* W_b2h     = (const float*)d_in[5];
  const float* b_b2h     = (const float*)d_in[6];
  const float* W_h2b     = (const float*)d_in[7];
  const float* b_h2b     = (const float*)d_in[8];
  const float* body_mask = (const float*)d_in[9];

  char* ws = (char*)d_ws;
  __hip_bfloat16* Wcat = (__hip_bfloat16*)(ws + WS_WCAT_OFF);
  __hip_bfloat16* Wb   = (__hip_bfloat16*)(ws + WS_WB_OFF);
  float*          bias = (float*)(ws + WS_BIAS_OFF);

  float* out        = (float*)d_out;
  float* hidden_out = out;                                           // [T,B,H]
  float* body_out   = hidden_out + (size_t)T_STEPS * BATCH * H_DIM;  // [T,B,BODY]
  float* h_fin      = body_out + (size_t)T_STEPS * BATCH * BODY_DIM; // [B,H]

  ctrnn_prep_kernel<<<512, 256, 0, stream>>>(
      W_i2h, b_i2h, W_h2h, b_h2h, W_b2h, b_b2h, W_h2b, Wcat, Wb, bias);

  ctrnn_scan_kernel<<<NUM_WG, THREADS, 0, stream>>>(
      x, Wcat, Wb, bias, b_h2b, body_mask, hidden_out, body_out, h_fin);
}
